// HybridQwen3MoeModel_16690242912524
// MI455X (gfx1250) — matrix operand order repaired and checked by reading, compiled, not run on hardware
//
#include <hip/hip_runtime.h>
#include <hip/hip_bf16.h>

// Problem constants (fixed by setup_inputs): T=4096, d=2048, E=64, I=768, K=8, C=1024
#define TT 4096
#define DD 2048
#define EE 64
#define II 768
#define KK 8
#define CC 1024

typedef __attribute__((ext_vector_type(16))) __bf16 v16bf;
typedef __attribute__((ext_vector_type(8)))  float  v8f;
typedef __attribute__((ext_vector_type(4)))  int    i32x4;
typedef __attribute__((ext_vector_type(8)))  int    i32x8;

union Frag16 { uint4 u[2]; v16bf v; };

__device__ __forceinline__ unsigned pack_bf16(float a, float b) {
  unsigned short ua = __builtin_bit_cast(unsigned short, (__bf16)a);
  unsigned short ub = __builtin_bit_cast(unsigned short, (__bf16)b);
  return (unsigned)ua | ((unsigned)ub << 16);   // lowers to v_cvt_pk_bf16_f32
}

// ---------------------------------------------------------------- zero output
__global__ void zero_out_kernel(float4* __restrict__ p, int n4) {
  int i = blockIdx.x * blockDim.x + threadIdx.x;
  if (i < n4) p[i] = make_float4(0.f, 0.f, 0.f, 0.f);
}

// ---------------------------------------------------------------- router
// one block (64 threads) per token: logits, softmax, top-8, renorm
__global__ __launch_bounds__(64)
void router_kernel(const float* __restrict__ hs, const float* __restrict__ rw,
                   int* __restrict__ topi, float* __restrict__ topw) {
  __shared__ float hrow[DD];
  __shared__ float red[64];
  __shared__ int   redi[64];
  __shared__ float selw[KK];
  __shared__ int   seli[KK];
  const int t = blockIdx.x, tid = threadIdx.x;

  for (int i = tid; i < DD; i += 64) hrow[i] = hs[(size_t)t * DD + i];
  __syncthreads();

  float acc = 0.f;
  for (int k = 0; k < DD; ++k) acc = fmaf(hrow[k], rw[k * EE + tid], acc);

  // softmax over 64 experts
  red[tid] = acc; __syncthreads();
  for (int s = 32; s > 0; s >>= 1) { if (tid < s) red[tid] = fmaxf(red[tid], red[tid + s]); __syncthreads(); }
  float mx = red[0]; __syncthreads();
  float ex = __expf(acc - mx);
  red[tid] = ex; __syncthreads();
  for (int s = 32; s > 0; s >>= 1) { if (tid < s) red[tid] += red[tid + s]; __syncthreads(); }
  float denom = red[0]; __syncthreads();
  float prob = ex / denom;

  // top-8 by repeated argmax (ties -> lowest index, matching stable top_k)
  float myp = prob;
  for (int r = 0; r < KK; ++r) {
    red[tid] = myp; redi[tid] = tid; __syncthreads();
    for (int s = 32; s > 0; s >>= 1) {
      if (tid < s) {
        if (red[tid + s] > red[tid]) { red[tid] = red[tid + s]; redi[tid] = redi[tid + s]; }
      }
      __syncthreads();
    }
    if (tid == 0) { selw[r] = red[0]; seli[r] = redi[0]; }
    __syncthreads();
    if (tid == seli[r]) myp = -1.f;
    __syncthreads();
  }
  if (tid < KK) {
    float s = 0.f;
    for (int r = 0; r < KK; ++r) s += selw[r];
    topi[(size_t)t * KK + tid] = seli[tid];
    topw[(size_t)t * KK + tid] = selw[tid] / s;
  }
}

// ---------------------------------------------------------------- assignment
// one block per expert; stable compaction in flat (t,k) order == reference argsort
__global__ __launch_bounds__(256)
void assign_kernel(const int* __restrict__ topi, const float* __restrict__ topw,
                   int* __restrict__ slot_token, float* __restrict__ slot_w) {
  const int e = blockIdx.x, tid = threadIdx.x;
  __shared__ int scan[256];
  __shared__ int basec;
  for (int i = tid; i < CC; i += 256) { slot_token[e * CC + i] = -1; slot_w[e * CC + i] = 0.f; }
  if (tid == 0) basec = 0;
  __syncthreads();
  for (int base = 0; base < TT * KK; base += 256) {
    int j = base + tid;
    int m = (topi[j] == e) ? 1 : 0;
    scan[tid] = m; __syncthreads();
    for (int off = 1; off < 256; off <<= 1) {
      int v = (tid >= off) ? scan[tid - off] : 0;
      __syncthreads();
      scan[tid] += v;
      __syncthreads();
    }
    int excl = scan[tid] - m;
    int total = scan[255];
    int pos = basec + excl;
    if (m && pos < CC) {
      slot_token[e * CC + pos] = j >> 3;     // token index
      slot_w[e * CC + pos]     = topw[j];    // gate weight
    }
    __syncthreads();
    if (tid == 0) basec += total;
    __syncthreads();
  }
}

// ---------------------------------------------------------------- GEMM1: h = silu(xg@w1) * (xg@w3)
// block tile 128M x 64N, BK=32; 8 waves as 4Mx2N, wave tile 32x32 for g and u.
// Double-buffered LDS pipeline: next tile staged in registers during WMMA.
__global__ __launch_bounds__(256)
void gemm1_kernel(const float* __restrict__ hs, const float* __restrict__ w1,
                  const float* __restrict__ w3, const int* __restrict__ slot_token,
                  __bf16* __restrict__ hbuf) {
  const int e  = blockIdx.z;
  const int m0 = blockIdx.y * 128;
  const int n0 = blockIdx.x * 64;
  const int tid = threadIdx.x;
  const int lane = tid & 31, wave = tid >> 5;
  const int waveM = wave & 3, waveN = wave >> 2;

  __shared__ __bf16 As[2][128 * 40];        // row-major, stride 40 halves (80B rows)
  __shared__ __bf16 Bs[2][2][64 * 40];      // [buf][mat], N-major (transposed)

  v8f accg[2][2] = {};
  v8f accu[2][2] = {};

  const float* w1e = w1 + (size_t)e * DD * II;
  const float* w3e = w3 + (size_t)e * DD * II;
  const int*   st  = slot_token + e * CC;

  float4 aReg[4];
  float4 bRa[2], bRb[2];
  const int bkk = (tid >> 4) * 2;           // 0,2,..,30
  const int bng = (tid & 15) * 4;           // 0..60

  auto loadA = [&](int k0) {
    #pragma unroll
    for (int i = 0; i < 4; ++i) {
      int v = tid + i * 256;
      int row = v >> 3, cg = (v & 7) * 4;
      int token = st[m0 + row];
      aReg[i] = make_float4(0.f, 0.f, 0.f, 0.f);
      if (token >= 0) aReg[i] = *(const float4*)(hs + (size_t)token * DD + k0 + cg);
    }
  };
  auto storeA = [&](int buf) {
    #pragma unroll
    for (int i = 0; i < 4; ++i) {
      int v = tid + i * 256;
      int row = v >> 3, cg = (v & 7) * 4;
      uint2 p;
      p.x = pack_bf16(aReg[i].x, aReg[i].y);
      p.y = pack_bf16(aReg[i].z, aReg[i].w);
      *(uint2*)&As[buf][row * 40 + cg] = p;
    }
  };
  auto loadB = [&](int k0) {
    #pragma unroll
    for (int mat = 0; mat < 2; ++mat) {
      const float* wp = mat ? w3e : w1e;
      bRa[mat] = *(const float4*)(wp + (size_t)(k0 + bkk) * II + n0 + bng);
      bRb[mat] = *(const float4*)(wp + (size_t)(k0 + bkk + 1) * II + n0 + bng);
    }
  };
  auto storeB = [&](int buf) {
    #pragma unroll
    for (int mat = 0; mat < 2; ++mat) {
      *(unsigned*)&Bs[buf][mat][(bng + 0) * 40 + bkk] = pack_bf16(bRa[mat].x, bRb[mat].x);
      *(unsigned*)&Bs[buf][mat][(bng + 1) * 40 + bkk] = pack_bf16(bRa[mat].y, bRb[mat].y);
      *(unsigned*)&Bs[buf][mat][(bng + 2) * 40 + bkk] = pack_bf16(bRa[mat].z, bRb[mat].z);
      *(unsigned*)&Bs[buf][mat][(bng + 3) * 40 + bkk] = pack_bf16(bRa[mat].w, bRb[mat].w);
    }
  };
  auto compute = [&](int buf) {
    Frag16 af[2];
    #pragma unroll
    for (int mt = 0; mt < 2; ++mt) {
      int row = waveM * 32 + mt * 16 + (lane & 15);
      int kh  = (lane >> 4) * 8;            // A: lanes 0-15 K{0..7,16..23}, lanes 16-31 K{8..15,24..31}
      af[mt].u[0] = *(const uint4*)&As[buf][row * 40 + kh];
      af[mt].u[1] = *(const uint4*)&As[buf][row * 40 + kh + 16];
    }
    Frag16 b1f[2], b3f[2];
    #pragma unroll
    for (int nt = 0; nt < 2; ++nt) {
      int col = waveN * 32 + nt * 16 + (lane & 15);
      int kh  = (lane >> 4) * 8;            // B: lanes 0-15 K{0..7,16..23}, lanes 16-31 K{8..15,24..31}
      const uint4* p1 = (const uint4*)&Bs[buf][0][col * 40 + kh];
      b1f[nt].u[0] = p1[0]; b1f[nt].u[1] = p1[2];
      const uint4* p3 = (const uint4*)&Bs[buf][1][col * 40 + kh];
      b3f[nt].u[0] = p3[0]; b3f[nt].u[1] = p3[2];
    }
    #pragma unroll
    for (int mt = 0; mt < 2; ++mt)
      #pragma unroll
      for (int nt = 0; nt < 2; ++nt) {
        accg[mt][nt] = __builtin_amdgcn_wmma_f32_16x16x32_bf16(
            false, af[mt].v, false, b1f[nt].v, (short)0, accg[mt][nt], false, false);
        accu[mt][nt] = __builtin_amdgcn_wmma_f32_16x16x32_bf16(
            false, af[mt].v, false, b3f[nt].v, (short)0, accu[mt][nt], false, false);
      }
  };

  // pipeline prologue
  loadA(0); loadB(0);
  storeA(0); storeB(0);
  __syncthreads();

  int buf = 0;
  for (int k0 = 0; k0 < DD; k0 += 32, buf ^= 1) {
    const int nk = k0 + 32;
    const bool more = nk < DD;
    if (more) {
      if (nk + 32 < DD) {  // keep the prefetch stream one tile ahead of the staged loads
        __builtin_prefetch(w1e + (size_t)(nk + 32 + (tid >> 4)) * II + n0 + (tid & 15) * 4, 0, 1);
        __builtin_prefetch(w3e + (size_t)(nk + 32 + (tid >> 4)) * II + n0 + (tid & 15) * 4, 0, 1);
      }
      loadA(nk); loadB(nk);              // stage next tile in registers
    }
    compute(buf);                        // WMMA on current LDS buffer
    if (more) { storeA(buf ^ 1); storeB(buf ^ 1); }
    __syncthreads();
  }

  // epilogue: h = silu(g)*u -> bf16 scratch
  #pragma unroll
  for (int mt = 0; mt < 2; ++mt)
    #pragma unroll
    for (int nt = 0; nt < 2; ++nt)
      #pragma unroll
      for (int r = 0; r < 8; ++r) {
        int m = m0 + waveM * 32 + mt * 16 + r + ((lane < 16) ? 0 : 8);
        int n = n0 + waveN * 32 + nt * 16 + (lane & 15);
        float g = accg[mt][nt][r], u = accu[mt][nt][r];
        float hv = (g / (1.f + __expf(-g))) * u;
        hbuf[((size_t)e * CC + m) * II + n] = (__bf16)hv;
      }
}

// ---------------------------------------------------------------- GEMM2: yo = h @ w2, scaled scatter-add
// block tile 128M x 128N, BK=32; 8 waves as 4Mx2N, wave tile 32x64.
// A-tile (bf16, contiguous in scratch) is fetched by the Tensor Data Mover into the
// *next* LDS buffer while the current one is consumed; wave 0 issues the TDM at the
// top of the iteration and only drains TENSORcnt right before the publishing barrier.
__global__ __launch_bounds__(256)
void gemm2_kernel(const __bf16* __restrict__ hbuf, const float* __restrict__ w2,
                  const int* __restrict__ slot_token, const float* __restrict__ slot_w,
                  float* __restrict__ out) {
  const int e  = blockIdx.z;
  const int m0 = blockIdx.y * 128;
  const int n0 = blockIdx.x * 128;
  const int tid = threadIdx.x;
  const int lane = tid & 31, wave = tid >> 5;
  const int waveM = wave & 3, waveN = wave >> 2;

  __shared__ __bf16 As[2][128 * 40];
  __shared__ __bf16 Bs[2][128 * 40];

  v8f acc[2][4] = {};

  const __bf16* he  = hbuf + (size_t)e * CC * II;
  const float*  w2e = w2 + (size_t)e * II * DD;

  float4 bRa[2], bRb[2];

  // TDM issue: D# reproduces the stride-40-half padded LDS layout
  auto tdmA = [&](int k0, int buf) {
    unsigned long long ga = (unsigned long long)(uintptr_t)(he + (size_t)m0 * II + k0);
    unsigned lds = (unsigned)(uintptr_t)&As[buf][0];   // flat addr[31:0] == group-segment offset
    i32x4 g0;
    g0[0] = 1;                                   // count=1, user descriptor
    g0[1] = (int)lds;                            // lds_addr
    g0[2] = (int)(unsigned)ga;                   // global_addr[31:0]
    g0[3] = (int)((unsigned)(ga >> 32) | (2u << 30));  // global_addr[56:32] | type=2
    i32x8 g1;
    g1[0] = (1 << 16)        // data_size = 2 bytes
          | (1 << 20)        // pad_enable
          | (3 << 22)        // pad_interval: 16 DWORDs (64B = one 32-elem bf16 row)
          | (3 << 25);       // pad_amount: 4 DWORDs (16B) -> 80B row stride
    g1[1] = (II & 0xffff) << 16;                 // tensor_dim0[15:0] = 768
    g1[2] = ((II >> 16) & 0xffff) | ((CC & 0xffff) << 16);   // dim0 hi | tensor_dim1 = 1024
    g1[3] = ((CC >> 16) & 0xffff) | (32 << 16);  // dim1 hi | tile_dim0 = 32
    g1[4] = 128;                                 // tile_dim1 = 128, tile_dim2 = 0
    g1[5] = II;                                  // tensor_dim0_stride = 768 elements
    g1[6] = 0;
    g1[7] = 0;
    asm volatile("tensor_load_to_lds %0, %1" :: "s"(g0), "s"(g1) : "memory");
  };
  auto loadB = [&](int k0) {
    #pragma unroll
    for (int i = 0; i < 2; ++i) {
      int v = tid + i * 256;
      int kk = (v >> 5) * 2, ng = (v & 31) * 4;
      bRa[i] = *(const float4*)(w2e + (size_t)(k0 + kk) * DD + n0 + ng);
      bRb[i] = *(const float4*)(w2e + (size_t)(k0 + kk + 1) * DD + n0 + ng);
    }
  };
  auto storeB = [&](int buf) {
    #pragma unroll
    for (int i = 0; i < 2; ++i) {
      int v = tid + i * 256;
      int kk = (v >> 5) * 2, ng = (v & 31) * 4;
      *(unsigned*)&Bs[buf][(ng + 0) * 40 + kk] = pack_bf16(bRa[i].x, bRb[i].x);
      *(unsigned*)&Bs[buf][(ng + 1) * 40 + kk] = pack_bf16(bRa[i].y, bRb[i].y);
      *(unsigned*)&Bs[buf][(ng + 2) * 40 + kk] = pack_bf16(bRa[i].z, bRb[i].z);
      *(unsigned*)&Bs[buf][(ng + 3) * 40 + kk] = pack_bf16(bRa[i].w, bRb[i].w);
    }
  };
  auto compute = [&](int buf) {
    Frag16 af[2];
    #pragma unroll
    for (int mt = 0; mt < 2; ++mt) {
      int row = waveM * 32 + mt * 16 + (lane & 15);
      int kh  = (lane >> 4) * 8;
      af[mt].u[0] = *(const uint4*)&As[buf][row * 40 + kh];
      af[mt].u[1] = *(const uint4*)&As[buf][row * 40 + kh + 16];
    }
    Frag16 bfr[4];
    #pragma unroll
    for (int nt = 0; nt < 4; ++nt) {
      int col = waveN * 64 + nt * 16 + (lane & 15);
      int kh  = (lane >> 4) * 8;
      const uint4* p = (const uint4*)&Bs[buf][col * 40 + kh];
      bfr[nt].u[0] = p[0]; bfr[nt].u[1] = p[2];
    }
    #pragma unroll
    for (int mt = 0; mt < 2; ++mt)
      #pragma unroll
      for (int nt = 0; nt < 4; ++nt)
        acc[mt][nt] = __builtin_amdgcn_wmma_f32_16x16x32_bf16(
            false, af[mt].v, false, bfr[nt].v, (short)0, acc[mt][nt], false, false);
  };

  // pipeline prologue: TDM A(0) + manual B(0)
  if (wave == 0) tdmA(0, 0);
  loadB(0); storeB(0);
  if (wave == 0) __builtin_amdgcn_s_wait_tensorcnt(0);
  __syncthreads();

  int buf = 0;
  for (int k0 = 0; k0 < II; k0 += 32, buf ^= 1) {
    const int nk = k0 + 32;
    const bool more = nk < II;
    if (more) {
      if (wave == 0) tdmA(nk, buf ^ 1);    // DMA next A tile while we compute
      if (nk + 32 < II)
        __builtin_prefetch(w2e + (size_t)(nk + 32 + (tid >> 5)) * DD + n0 + (tid & 31) * 4, 0, 1);
      loadB(nk);                           // stage next B tile in registers
    }
    compute(buf);
    if (more) {
      storeB(buf ^ 1);
      if (wave == 0) __builtin_amdgcn_s_wait_tensorcnt(0);  // TDM had a full compute stage
    }
    __syncthreads();
  }

  const int*   st = slot_token + e * CC;
  const float* sw = slot_w + e * CC;
  #pragma unroll
  for (int mt = 0; mt < 2; ++mt) {
    int mbase = waveM * 32 + mt * 16 + ((lane < 16) ? 0 : 8);
    #pragma unroll
    for (int r = 0; r < 8; ++r) {
      int slot = m0 + mbase + r;
      int token = st[slot];
      if (token < 0) continue;
      float wgt = sw[slot];
      #pragma unroll
      for (int nt = 0; nt < 4; ++nt) {
        int n = n0 + waveN * 64 + nt * 16 + (lane & 15);
        unsafeAtomicAdd(&out[(size_t)token * DD + n], acc[mt][nt][r] * wgt);
      }
    }
  }
}

// ---------------------------------------------------------------- launcher
extern "C" void kernel_launch(void* const* d_in, const int* in_sizes, int n_in,
                              void* d_out, int out_size, void* d_ws, size_t ws_size,
                              hipStream_t stream) {
  const float* hs = (const float*)d_in[0];      // (T, d)
  const float* rw = (const float*)d_in[1];      // (d, E)
  const float* w1 = (const float*)d_in[2];      // (E, d, I)
  const float* w3 = (const float*)d_in[3];      // (E, d, I)
  const float* w2 = (const float*)d_in[4];      // (E, I, d)
  (void)d_in[5]; (void)in_sizes; (void)n_in; (void)ws_size; (void)out_size;
  float* out = (float*)d_out;

  // workspace layout (all 256B aligned)
  char* ws = (char*)d_ws;
  int*    topi       = (int*)(ws);                         // T*K ints    = 128 KB
  float*  topw       = (float*)(ws + 131072);              // T*K floats  = 128 KB
  int*    slot_token = (int*)(ws + 262144);                // E*C ints    = 256 KB
  float*  slot_w     = (float*)(ws + 524288);              // E*C floats  = 256 KB
  __bf16* hbuf       = (__bf16*)(ws + 786432);             // E*C*I bf16  = 96 MB

  // 1) zero the output (we accumulate with atomics)
  {
    int n4 = (TT * DD) / 4;
    zero_out_kernel<<<(n4 + 255) / 256, 256, 0, stream>>>((float4*)out, n4);
  }
  // 2) router: logits -> softmax -> top-8 -> renorm
  router_kernel<<<TT, 64, 0, stream>>>(hs, rw, topi, topw);
  // 3) deterministic expert assignment with capacity drop
  assign_kernel<<<EE, 256, 0, stream>>>(topi, topw, slot_token, slot_w);
  // 4) h = silu(xg@w1) * (xg@w3)   (gathered A, double-buffered bf16 WMMA pipeline)
  gemm1_kernel<<<dim3(II / 64, CC / 128, EE), 256, 0, stream>>>(hs, w1, w3, slot_token, hbuf);
  // 5) out += weight * (h @ w2)    (TDM-fed double-buffered bf16 WMMA, scaled atomic scatter)
  gemm2_kernel<<<dim3(DD / 128, CC / 128, EE), 256, 0, stream>>>(hbuf, w2, slot_token, slot_w, out);
}
